// LightningAttention_45131516346421
// MI455X (gfx1250) — matrix operand order repaired and checked by reading, not compiled, not run on hardware
//
#include <hip/hip_runtime.h>
#include <hip/hip_bf16.h>
#include <math.h>

typedef __attribute__((ext_vector_type(16))) _Float16 v16h;
typedef __attribute__((ext_vector_type(8)))  _Float16 v8h;
typedef __attribute__((ext_vector_type(8)))  float    v8f;

#define DIM   2048
#define HDIM  128
#define MTOK  16384     // B*L
#define EPS   1e-6f

#if __has_builtin(__builtin_amdgcn_global_load_async_to_lds_b128) && \
    __has_builtin(__builtin_amdgcn_s_wait_asynccnt)
#define USE_ASYNC_LDS 1
#else
#define USE_ASYNC_LDS 0
#endif

// exact parameter type from the hipcc diagnostic: int vector_size(16), AS(1)/AS(3)
typedef int v4i __attribute__((vector_size(16)));
typedef __attribute__((address_space(1))) v4i* gv4i_p;
typedef __attribute__((address_space(3))) v4i* lv4i_p;

__device__ __forceinline__ gv4i_p as_gv4(const void* p) {
  return (gv4i_p)(unsigned long long)(uintptr_t)p;  // generic == AS1 numerically
}
__device__ __forceinline__ lv4i_p as_lv4(void* p) {
  return (lv4i_p)(unsigned int)(uintptr_t)p;        // low 32 bits == LDS offset
}

// ---------------------------------------------------------------------------
// Kernel 1: RMSNorm(x, w_in) and cast to f16.   one block per token
// ---------------------------------------------------------------------------
__global__ __launch_bounds__(256)
void k_rmsnorm_cast(const float* __restrict__ x, const float* __restrict__ w,
                    _Float16* __restrict__ xnh) {
  __shared__ float red[256];
  const int t   = blockIdx.x;
  const int tid = threadIdx.x;
  const float* xp = x + (size_t)t * DIM;
  float xv[8];
  float ss = 0.f;
#pragma unroll
  for (int i = 0; i < 8; ++i) { xv[i] = xp[i * 256 + tid]; ss += xv[i] * xv[i]; }
  red[tid] = ss;
  __syncthreads();
  for (int off = 128; off > 0; off >>= 1) {
    if (tid < off) red[tid] += red[tid + off];
    __syncthreads();
  }
  const float rms = rsqrtf(red[0] * (1.0f / DIM) + EPS);
  _Float16* op = xnh + (size_t)t * DIM;
#pragma unroll
  for (int i = 0; i < 8; ++i) {
    const int idx = i * 256 + tid;
    op[idx] = (_Float16)(xv[i] * rms * w[idx]);
  }
}

// ---------------------------------------------------------------------------
// Kernel 2: cast 4 weight matrices f32 -> f16 (concatenated in workspace)
// ---------------------------------------------------------------------------
__global__ __launch_bounds__(256)
void k_cast_w(const float* __restrict__ wq, const float* __restrict__ wk,
              const float* __restrict__ wv, const float* __restrict__ wg,
              _Float16* __restrict__ wh) {
  const size_t DD = (size_t)DIM * DIM;
  const size_t n  = 4 * DD;
  size_t i = (size_t)blockIdx.x * blockDim.x + threadIdx.x;
  const size_t stride = (size_t)gridDim.x * blockDim.x;
  for (; i < n; i += stride) {
    const size_t w = i / DD, r = i - w * DD;
    const float* src = (w == 0) ? wq : (w == 1) ? wk : (w == 2) ? wv : wg;
    wh[i] = (_Float16)src[r];
  }
}

// ---------------------------------------------------------------------------
// Kernel 3: Y[w] = maybe_silu( xn @ W[w]^T ), f16 in / f16 out, f32 accum.
// Tile: 128(M) x 128(N) x 64(K).  8 waves (2x4), each wave: 4x2 wmma tiles.
// Tiles staged via GLOBAL_LOAD_ASYNC_TO_LDS_B128 (no VGPR staging, ASYNCcnt).
// ---------------------------------------------------------------------------
#define TM  128
#define TN  128
#define TK  64
#define LDP 72   // padded LDS row length in f16 (144B -> conflict-free)

__global__ __launch_bounds__(256)
void k_gemm_silu(const _Float16* __restrict__ xnh,
                 const _Float16* __restrict__ wh,
                 _Float16* __restrict__ yh) {
  __shared__ __align__(16) _Float16 As[2][TM][LDP];
  __shared__ __align__(16) _Float16 Bs[2][TN][LDP];

  const int tid   = threadIdx.x;
  const int lane  = tid & 31;
  const int wave  = tid >> 5;
  const int waveM = wave >> 2;   // 0..1  (64 rows each)
  const int waveN = wave & 3;    // 0..3  (32 cols each)

  const int w  = blockIdx.x >> 4;          // which weight matrix (0..3)
  const int n0 = (blockIdx.x & 15) * TN;   // col offset inside weight
  const int m0 = blockIdx.y * TM;

  const _Float16* Ag = xnh + (size_t)m0 * DIM;
  const _Float16* Bg = wh + (size_t)w * DIM * DIM + (size_t)n0 * DIM;

  const int ldr = tid >> 3;        // 0..31 : row within a load pass
  const int ldc = (tid & 7) * 8;   // f16 column (16B chunks)

#if USE_ASYNC_LDS
  // ---- issue one tile's worth of async copies (4x16B A + 4x16B B per thread)
  auto stage_tile = [&](int buf, int k0) {
#pragma unroll
    for (int p = 0; p < 4; ++p) {
      __builtin_amdgcn_global_load_async_to_lds_b128(
          as_gv4(Ag + (size_t)(ldr + p * 32) * DIM + k0 + ldc),
          as_lv4(&As[buf][ldr + p * 32][ldc]), 0, 0);
      __builtin_amdgcn_global_load_async_to_lds_b128(
          as_gv4(Bg + (size_t)(ldr + p * 32) * DIM + k0 + ldc),
          as_lv4(&Bs[buf][ldr + p * 32][ldc]), 0, 0);
    }
  };
  stage_tile(0, 0);
  __builtin_amdgcn_s_wait_asynccnt(0);
  __syncthreads();
#else
  {
    float4 aReg[4], bReg[4];
#pragma unroll
    for (int p = 0; p < 4; ++p) {
      aReg[p] = *(const float4*)(Ag + (size_t)(ldr + p * 32) * DIM + ldc);
      bReg[p] = *(const float4*)(Bg + (size_t)(ldr + p * 32) * DIM + ldc);
    }
#pragma unroll
    for (int p = 0; p < 4; ++p) {
      *(float4*)(&As[0][ldr + p * 32][ldc]) = aReg[p];
      *(float4*)(&Bs[0][ldr + p * 32][ldc]) = bReg[p];
    }
    __syncthreads();
  }
#endif

  v8f acc[4][2] = {};
  const int arow = waveM * 64 + (lane & 15);
  const int brow = waveN * 32 + (lane & 15);
  const int hi   = lane >> 4;

  for (int kt = 0; kt < 32; ++kt) {
    const int cur = kt & 1;

#if USE_ASYNC_LDS
    if (kt < 31) stage_tile(1 - cur, (kt + 1) * TK);
#else
    float4 aReg[4], bReg[4];
    if (kt < 31) {
      const int k0 = (kt + 1) * TK;
#pragma unroll
      for (int p = 0; p < 4; ++p) {
        aReg[p] = *(const float4*)(Ag + (size_t)(ldr + p * 32) * DIM + k0 + ldc);
        bReg[p] = *(const float4*)(Bg + (size_t)(ldr + p * 32) * DIM + k0 + ldc);
      }
    }
#endif
    if (kt < 30) {  // warm L2 for tile kt+2 (global_prefetch_b8)
      const int k2 = (kt + 2) * TK;
      __builtin_prefetch(Ag + (size_t)ldr * DIM + k2 + ldc, 0, 1);
      __builtin_prefetch(Bg + (size_t)ldr * DIM + k2 + ldc, 0, 1);
    }

#pragma unroll
    for (int ks = 0; ks < TK; ks += 32) {
      v16h afrag[4], bfrag[2];
#pragma unroll
      for (int mi = 0; mi < 4; ++mi) {
        const _Float16* ap = &As[cur][arow + mi * 16][ks + hi * 8];
        v8h l0 = *(const v8h*)ap;
        v8h l1 = *(const v8h*)(ap + 16);
#pragma unroll
        for (int i = 0; i < 8; ++i) { afrag[mi][i] = l0[i]; afrag[mi][i + 8] = l1[i]; }
      }
#pragma unroll
      for (int ni = 0; ni < 2; ++ni) {
        const _Float16* bp = &Bs[cur][brow + ni * 16][ks + hi * 8];
        v8h l0 = *(const v8h*)bp;
        v8h l1 = *(const v8h*)(bp + 16);
#pragma unroll
        for (int i = 0; i < 8; ++i) { bfrag[ni][i] = l0[i]; bfrag[ni][i + 8] = l1[i]; }
      }
#pragma unroll
      for (int mi = 0; mi < 4; ++mi)
#pragma unroll
        for (int ni = 0; ni < 2; ++ni)
          acc[mi][ni] = __builtin_amdgcn_wmma_f32_16x16x32_f16(
              false, afrag[mi], false, bfrag[ni], (short)0, acc[mi][ni],
              false, false);
    }

    if (kt < 31) {
#if USE_ASYNC_LDS
      __builtin_amdgcn_s_wait_asynccnt(0);
      __syncthreads();
#else
#pragma unroll
      for (int p = 0; p < 4; ++p) {
        *(float4*)(&As[1 - cur][ldr + p * 32][ldc]) = aReg[p];
        *(float4*)(&Bs[1 - cur][ldr + p * 32][ldc]) = bReg[p];
      }
      __syncthreads();
#endif
    }
  }

  // epilogue: optional SiLU (q,k,v only), cast to f16, store
  const bool dosilu = (w < 3);
  _Float16* yp = yh + (size_t)w * MTOK * DIM;
#pragma unroll
  for (int mi = 0; mi < 4; ++mi) {
#pragma unroll
    for (int ni = 0; ni < 2; ++ni) {
      const int col = n0 + waveN * 32 + ni * 16 + (lane & 15);
#pragma unroll
      for (int r = 0; r < 8; ++r) {
        const int row = m0 + waveM * 64 + mi * 16 + r + hi * 8;
        float v = acc[mi][ni][r];
        if (dosilu) v = v / (1.f + __expf(-v));
        yp[(size_t)row * DIM + col] = (_Float16)v;
      }
    }
  }
}

// ---------------------------------------------------------------------------
// Kernel 4: per-token 16x16 head attention + gate + output RMSNorm.
// One wave per token (8 tokens per 256-thread block).
// Scores via v_wmma (A=q 16x128, B=k^T), softmax across lanes (wave32 shfl).
// ---------------------------------------------------------------------------
__global__ __launch_bounds__(256)
void k_attn(const _Float16* __restrict__ yh, const float* __restrict__ wout,
            float* __restrict__ out) {
  __shared__ float               sc[8][16][16];
  __shared__ __align__(16) _Float16 vsh[8][16 * HDIM];

  const int tid  = threadIdx.x;
  const int lane = tid & 31;
  const int wsl  = tid >> 5;  // wave slot = token slot
  const size_t tok = (size_t)blockIdx.x * 8 + wsl;
  const size_t MD  = (size_t)MTOK * DIM;

  const _Float16* qp = yh + tok * DIM;
  const _Float16* kp = yh + MD + tok * DIM;
  const _Float16* vp = yh + 2 * MD + tok * DIM;
  const _Float16* gp = yh + 3 * MD + tok * DIM;

  const int hi   = lane >> 4;
  const int lo16 = lane & 15;

  // stage this token's V (16x128 f16) into LDS
#pragma unroll
  for (int p = 0; p < 8; ++p)
    *(v8h*)(&vsh[wsl][p * 256 + lane * 8]) = *(const v8h*)(vp + p * 256 + lane * 8);

  // scores[h][g] = sum_d q[h,d] k[g,d]    (4 x wmma, K=128)
  v8f s = {};
#pragma unroll
  for (int kk = 0; kk < 128; kk += 32) {
    v16h aq, bk;
    {
      const _Float16* ap = qp + lo16 * HDIM + kk + hi * 8;
      v8h l0 = *(const v8h*)ap;
      v8h l1 = *(const v8h*)(ap + 16);
#pragma unroll
      for (int i = 0; i < 8; ++i) { aq[i] = l0[i]; aq[i + 8] = l1[i]; }
    }
    {
      const _Float16* bp = kp + lo16 * HDIM + kk + hi * 8;
      v8h l0 = *(const v8h*)bp;
      v8h l1 = *(const v8h*)(bp + 16);
#pragma unroll
      for (int i = 0; i < 8; ++i) { bk[i] = l0[i]; bk[i + 8] = l1[i]; }
    }
    s = __builtin_amdgcn_wmma_f32_16x16x32_f16(false, aq, false, bk, (short)0,
                                               s, false, false);
  }

  // row-wise softmax: row lives across 16 lanes of a half-wave
  const float scale = 0.0883883476483184f;  // 1/sqrt(128)
#pragma unroll
  for (int r = 0; r < 8; ++r) {
    float v = s[r] * scale;
    float m = v;
#pragma unroll
    for (int mk = 1; mk <= 8; mk <<= 1) m = fmaxf(m, __shfl_xor(m, mk, 32));
    const float e = __expf(v - m);
    float sum = e;
#pragma unroll
    for (int mk = 1; mk <= 8; mk <<= 1) sum += __shfl_xor(sum, mk, 32);
    sc[wsl][r + hi * 8][lo16] = e / sum;
  }
  __syncthreads();

  // out[h,d] = sum_g attn[h,g] v[g,d]; gate with sigmoid(g); collect sum-sq
  float o[64];
  float ss = 0.f;
#pragma unroll
  for (int j = 0; j < 64; ++j) {
    const int idx = j * 32 + lane;
    const int h = idx >> 7, d = idx & 127;
    float acc = 0.f;
#pragma unroll
    for (int g = 0; g < 16; ++g)
      acc += sc[wsl][h][g] * (float)vsh[wsl][g * HDIM + d];
    const float gv = (float)gp[idx];
    acc *= 1.f / (1.f + __expf(-gv));
    o[j] = acc;
    ss += acc * acc;
  }
#pragma unroll
  for (int mk = 1; mk <= 16; mk <<= 1) ss += __shfl_xor(ss, mk, 32);
  const float rms = rsqrtf(ss * (1.0f / DIM) + EPS);

  float* op = out + tok * DIM;
#pragma unroll
  for (int j = 0; j < 64; ++j) {
    const int idx = j * 32 + lane;
    op[idx] = o[j] * rms * wout[idx];
  }
}

// ---------------------------------------------------------------------------
// Launch
// ---------------------------------------------------------------------------
extern "C" void kernel_launch(void* const* d_in, const int* in_sizes, int n_in,
                              void* d_out, int out_size, void* d_ws,
                              size_t ws_size, hipStream_t stream) {
  const float* x   = (const float*)d_in[0];
  const float* wq  = (const float*)d_in[1];
  const float* wk  = (const float*)d_in[2];
  const float* wv  = (const float*)d_in[3];
  const float* wg  = (const float*)d_in[4];
  const float* win = (const float*)d_in[5];
  const float* wou = (const float*)d_in[6];
  float* out = (float*)d_out;

  char* ws = (char*)d_ws;
  _Float16* xnh = (_Float16*)ws;                            // 64 MiB  (16384x2048 f16)
  _Float16* wh  = (_Float16*)(ws + (size_t)67108864);       // 32 MiB  (4x2048x2048 f16)
  _Float16* yh  = (_Float16*)(ws + (size_t)100663296);      // 256 MiB (q,k,v,g f16)

  k_rmsnorm_cast<<<MTOK, 256, 0, stream>>>(x, win, xnh);
  k_cast_w<<<4096, 256, 0, stream>>>(wq, wk, wv, wg, wh);
  k_gemm_silu<<<dim3(64, 128), 256, 0, stream>>>(xnh, wh, yh);
  k_attn<<<MTOK / 8, 256, 0, stream>>>(yh, wou, out);
}
